// SGDVAE_52175262712004
// MI455X (gfx1250) — matrix operand order repaired and checked by reading, not compiled, not run on hardware
//
#include <hip/hip_runtime.h>
#include <hip/hip_bf16.h>

#define NB 8
#define NN 512
#define ND 16
#define NH 50
#define NOUT 32

typedef _Float16 f16;
typedef __attribute__((ext_vector_type(16))) _Float16 v16h;
typedef __attribute__((ext_vector_type(8)))  _Float16 v8h;
typedef __attribute__((ext_vector_type(8)))  float    v8f;

__device__ __forceinline__ float lrelu(float v) { return v >= 0.f ? v : 0.05f * v; }

__device__ __forceinline__ float sq4(float4 a, float4 b) {
    float dx = a.x - b.x, dy = a.y - b.y, dz = a.z - b.z, dw = a.w - b.w;
    return dx * dx + dy * dy + dz * dz + dw * dw;
}

// ---------------------------------------------------------------------------
// P0: per (b,i) row: dis[i,:], f16 copies of dis/adj, deg[i]=sum_j adj,
//     adjdisjk[i]=sum_j adj*dis.   grid = B*N blocks x 256 threads
// ---------------------------------------------------------------------------
__global__ void k_prep0(const float* __restrict__ x, const float* __restrict__ adj,
                        f16* __restrict__ disH, f16* __restrict__ adjH,
                        float* __restrict__ deg, float* __restrict__ adjdisjk)
{
    int row = blockIdx.x;          // b*N + i
    int b   = row >> 9;
    int i   = row & (NN - 1);
    const float* xb = x + (size_t)b * NN * ND;
    const float4* xi4 = (const float4*)(xb + i * ND);
    float4 xi0 = xi4[0], xi1 = xi4[1], xi2 = xi4[2], xi3 = xi4[3];

    const float* arow = adj + (size_t)row * NN;
    f16* drow  = disH + (size_t)row * NN;
    f16* ahrow = adjH + (size_t)row * NN;

    float sdeg = 0.f, sjk = 0.f;
    #pragma unroll
    for (int p = 0; p < 2; ++p) {
        int j = threadIdx.x + p * 256;
        const float4* xj4 = (const float4*)(xb + j * ND);
        float ss = 1e-10f;
        ss += sq4(xi0, xj4[0]);
        ss += sq4(xi1, xj4[1]);
        ss += sq4(xi2, xj4[2]);
        ss += sq4(xi3, xj4[3]);
        float dd = sqrtf(ss);
        float a  = arow[j];
        drow[j]  = (f16)dd;
        ahrow[j] = (f16)a;
        sdeg += a;
        sjk  += a * dd;
    }
    __shared__ float r1[256], r2[256];
    r1[threadIdx.x] = sdeg;
    r2[threadIdx.x] = sjk;
    __syncthreads();
    for (int s = 128; s > 0; s >>= 1) {
        if (threadIdx.x < s) {
            r1[threadIdx.x] += r1[threadIdx.x + s];
            r2[threadIdx.x] += r2[threadIdx.x + s];
        }
        __syncthreads();
    }
    if (threadIdx.x == 0) { deg[row] = r1[0]; adjdisjk[row] = r2[0]; }
}

// ---------------------------------------------------------------------------
// P1: per (b,n): x@Wx, x@Wy+b1, x@Wa+b2, x@W3[:d] (f32) and transposed,
//     zero-padded f16 copies of x@Wz / x@Wb for the WMMA GEMM.
//     grid = B*N/256 blocks x 256 threads
// ---------------------------------------------------------------------------
__global__ void k_prep1(const float* __restrict__ x,
                        const float* __restrict__ W1, const float* __restrict__ b1,
                        const float* __restrict__ W2, const float* __restrict__ b2,
                        const float* __restrict__ W3,
                        float* __restrict__ xWx, float* __restrict__ xWyb1,
                        float* __restrict__ xWab2, float* __restrict__ xW3,
                        f16* __restrict__ xWT)
{
    __shared__ float sW1[48 * 50 + 50];   // Wx|Wy|Wz rows + b1
    __shared__ float sW2[32 * 50 + 50];   // Wa|Wb rows + b2
    __shared__ float sW3[16 * 32];        // W3[:d]
    for (int t = threadIdx.x; t < 48 * 50; t += 256) sW1[t] = W1[t];
    for (int t = threadIdx.x; t < 50;      t += 256) sW1[48 * 50 + t] = b1[t];
    for (int t = threadIdx.x; t < 32 * 50; t += 256) sW2[t] = W2[t];
    for (int t = threadIdx.x; t < 50;      t += 256) sW2[32 * 50 + t] = b2[t];
    for (int t = threadIdx.x; t < 16 * 32; t += 256) sW3[t] = W3[t];
    __syncthreads();

    int row = blockIdx.x * 256 + threadIdx.x;   // b*N + n
    int b   = row >> 9;
    int n   = row & (NN - 1);

    float xv[ND];
    const float* xr = x + (size_t)row * ND;
    #pragma unroll
    for (int d = 0; d < ND; ++d) xv[d] = xr[d];

    for (int h = 0; h < 64; ++h) {
        float sz = 0.f, sb = 0.f;
        if (h < NH) {
            float sx = 0.f, sy = 0.f, sa = 0.f;
            #pragma unroll
            for (int d = 0; d < ND; ++d) {
                float xd = xv[d];
                sx += xd * sW1[d * 50 + h];
                sy += xd * sW1[(16 + d) * 50 + h];
                sz += xd * sW1[(32 + d) * 50 + h];
                sa += xd * sW2[d * 50 + h];
                sb += xd * sW2[(16 + d) * 50 + h];
            }
            xWx  [(size_t)row * 64 + h] = sx;
            xWyb1[(size_t)row * 64 + h] = sy + sW1[48 * 50 + h];
            xWab2[(size_t)row * 64 + h] = sa + sW2[32 * 50 + h];
        }
        xWT[((size_t)b * 128 + h)      * NN + n] = (f16)((h < NH) ? sz : 0.f);
        xWT[((size_t)b * 128 + 64 + h) * NN + n] = (f16)((h < NH) ? sb : 0.f);
    }
    #pragma unroll
    for (int o = 0; o < NOUT; ++o) {
        float s = 0.f;
        #pragma unroll
        for (int d = 0; d < ND; ++d) s += xv[d] * sW3[d * 32 + o];
        xW3[(size_t)row * 32 + o] = s;
    }
}

// ---------------------------------------------------------------------------
// WMMA NT GEMM: C[M x Nc](f32) = A[M x K](f16,row-major) * Bt[Nc x K]^T
// one wave per 16x16 output tile; v_wmma_f32_16x16x32_f16, f32 accumulate.
// ---------------------------------------------------------------------------
__global__ void k_gemm_nt(const f16* __restrict__ A, const f16* __restrict__ Bt,
                          float* __restrict__ C,
                          int tilesM, int tilesN, int K, int ldc,
                          long sA, long sB, long sC)
{
    int gwave = (blockIdx.x * blockDim.x + threadIdx.x) >> 5;
    int lane  = threadIdx.x & 31;
    int tpb   = tilesM * tilesN;
    int b     = gwave / tpb;
    int t     = gwave - b * tpb;
    int ti    = t / tilesN;
    int tj    = t - ti * tilesN;

    // A fragment: lane holds row (lane&15); K chunks {kb..kb+7, kb+16..kb+23}, kb = (lane>=16)?8:0
    const f16* pa = A + (size_t)b * sA
                      + (size_t)(ti * 16 + (lane & 15)) * K + ((lane >> 4) * 8);
    // B fragment: lane holds col (lane&15); K chunks {kb..kb+7, kb+16..kb+23}, kb = (lane>=16)?8:0
    const f16* pb = Bt + (size_t)b * sB
                       + (size_t)(tj * 16 + (lane & 15)) * K + ((lane >> 4) * 8);

    v8f acc = {};
    #pragma unroll 2
    for (int k0 = 0; k0 < K; k0 += 32) {
        v8h alo = *(const v8h*)(pa + k0);
        v8h ahi = *(const v8h*)(pa + k0 + 16);
        v16h av = __builtin_shufflevector(alo, ahi, 0,1,2,3,4,5,6,7,8,9,10,11,12,13,14,15);
        v8h blo = *(const v8h*)(pb + k0);
        v8h bhi = *(const v8h*)(pb + k0 + 16);
        v16h bv = __builtin_shufflevector(blo, bhi, 0,1,2,3,4,5,6,7,8,9,10,11,12,13,14,15);
        acc = __builtin_amdgcn_wmma_f32_16x16x32_f16(
                  false, av, false, bv, (short)0, acc, false, false);
    }
    // C layout: VGPR r -> row ((lane>=16)?8:0)+r, col lane&15
    float* pc = C + (size_t)b * sC
                  + (size_t)(ti * 16 + ((lane >> 4) << 3)) * ldc + (tj * 16 + (lane & 15));
    #pragma unroll
    for (int r = 0; r < 8; ++r) pc[(size_t)r * ldc] = acc[r];
}

// ---------------------------------------------------------------------------
// V: per-j vector v[j,h] for inner, and m2 base term base2[i,h]
// ---------------------------------------------------------------------------
__global__ void k_vj(const float* __restrict__ xWyb1, const float* __restrict__ xWab2,
                     const float* __restrict__ adjW,
                     const float* __restrict__ deg, const float* __restrict__ adjdisjk,
                     const float* __restrict__ W1, const float* __restrict__ W2,
                     float* __restrict__ vj, float* __restrict__ base2)
{
    int idx = blockIdx.x * 256 + threadIdx.x;  // over B*N*64
    int h   = idx & 63;
    int row = idx >> 6;                        // b*N + j
    if (h < NH) {
        float d = deg[row], s = adjdisjk[row];
        vj[(size_t)row * 64 + h] =
            d * xWyb1[(size_t)row * 64 + h] + adjW[(size_t)row * 128 + h]
            + s * W1[49 * 50 + h];
        base2[(size_t)row * 64 + h] =
            d * xWab2[(size_t)row * 64 + h] + adjW[(size_t)row * 128 + 64 + h]
            + s * W2[32 * 50 + h];
    }
}

// ---------------------------------------------------------------------------
// F: sparse m3 reduction: macc[i,h] = sum_j adj[i,j]*lrelu(adj[i,j]*inner[i,j,h])
//    block = 64 threads (h), one (b,i) row per block; row data staged in LDS
// ---------------------------------------------------------------------------
__global__ void k_m3(const f16* __restrict__ adjH, const f16* __restrict__ disH,
                     const float* __restrict__ adjdisik, const float* __restrict__ deg,
                     const float* __restrict__ xWx, const float* __restrict__ vj,
                     const float* __restrict__ W1, float* __restrict__ macc)
{
    int row = blockIdx.x;          // b*N + i
    int b   = row >> 9;
    int tid = threadIdx.x;         // 0..63

    __shared__ float sa[NN], sd[NN], sik[NN], sdg[NN];
    {
        const f16*  ar  = adjH + (size_t)row * NN;
        const f16*  dr  = disH + (size_t)row * NN;
        const float* ir = adjdisik + (size_t)row * NN;
        const float* gr = deg + (size_t)b * NN;
        v8h av = *(const v8h*)(ar + tid * 8);
        v8h dv = *(const v8h*)(dr + tid * 8);
        #pragma unroll
        for (int t = 0; t < 8; ++t) {
            sa[tid * 8 + t] = (float)av[t];
            sd[tid * 8 + t] = (float)dv[t];
        }
        #pragma unroll
        for (int t = 0; t < 8; ++t) {
            sik[tid * 8 + t] = ir[tid * 8 + t];
            sdg[tid * 8 + t] = gr[tid * 8 + t];
        }
    }
    __syncthreads();

    if (tid < NH) {
        float xwx = xWx[(size_t)row * 64 + tid];
        float wij = W1[48 * 50 + tid];
        float wik = W1[50 * 50 + tid];
        const float* vb = vj + (size_t)(b * NN) * 64 + tid;
        float acc = 0.f;
        for (int j = 0; j < NN; ++j) {
            float a = sa[j];
            if (a != 0.f) {
                float dj    = sdg[j];
                float inner = dj * xwx + vb[(size_t)j * 64]
                            + (sd[j] * dj) * wij + sik[j] * wik;
                acc += a * lrelu(a * inner);
            }
        }
        macc[(size_t)row * 64 + tid] = acc;
    }
}

// ---------------------------------------------------------------------------
// M: m2 = lrelu(base2 + macc@Wm); out = lrelu(xW3 + m2@W3[d:] + b3)
// ---------------------------------------------------------------------------
__global__ void k_m2m1(const float* __restrict__ base2, const float* __restrict__ macc,
                       const float* __restrict__ xW3, const float* __restrict__ W2,
                       const float* __restrict__ W3, const float* __restrict__ b3,
                       float* __restrict__ out)
{
    int row = blockIdx.x;
    int tid = threadIdx.x;   // 0..63
    __shared__ float sm[64];
    __shared__ float sm2[64];
    sm[tid] = (tid < NH) ? macc[(size_t)row * 64 + tid] : 0.f;
    __syncthreads();
    if (tid < NH) {
        float s = base2[(size_t)row * 64 + tid];
        for (int h0 = 0; h0 < NH; ++h0) s += sm[h0] * W2[(33 + h0) * 50 + tid];
        sm2[tid] = lrelu(s);
    }
    __syncthreads();
    if (tid < NOUT) {
        float o = xW3[(size_t)row * 32 + tid] + b3[tid];
        for (int hh = 0; hh < NH; ++hh) o += sm2[hh] * W3[(16 + hh) * 32 + tid];
        out[(size_t)row * 32 + tid] = lrelu(o);
    }
}

// ---------------------------------------------------------------------------
extern "C" void kernel_launch(void* const* d_in, const int* in_sizes, int n_in,
                              void* d_out, int out_size, void* d_ws, size_t ws_size,
                              hipStream_t stream)
{
    const float* x   = (const float*)d_in[0];
    const float* adj = (const float*)d_in[1];
    const float* W1  = (const float*)d_in[2];
    const float* b1  = (const float*)d_in[3];
    const float* W2  = (const float*)d_in[4];
    const float* b2  = (const float*)d_in[5];
    const float* W3  = (const float*)d_in[6];
    const float* b3  = (const float*)d_in[7];
    float* out = (float*)d_out;

    char* ws = (char*)d_ws;
    const size_t MB = 1u << 20;
    f16*   disH     = (f16*)  (ws);                       // 4 MB
    f16*   adjH     = (f16*)  (ws + 4 * MB);              // 4 MB
    float* adjdisik = (float*)(ws + 8 * MB);              // 8 MB
    f16*   xWT      = (f16*)  (ws + 16 * MB);             // 1 MB  [B][128][512]
    float* adjW     = (float*)(ws + 17 * MB);             // 2 MB  [B][512][128]
    float* deg      = (float*)(ws + 19 * MB);             // 16 KB
    float* adjdisjk = (float*)(ws + 19 * MB + 65536);     // 16 KB
    float* xWx      = (float*)(ws + 19 * MB + 131072);    // 1 MB  [B*N][64]
    float* xWyb1    = (float*)(ws + 20 * MB + 131072);    // 1 MB
    float* xWab2    = (float*)(ws + 21 * MB + 131072);    // 1 MB
    float* xW3      = (float*)(ws + 22 * MB + 131072);    // 0.5 MB
    float* vj       = (float*)(ws + 23 * MB);             // 1 MB
    float* base2    = (float*)(ws + 24 * MB);             // 1 MB
    float* macc     = (float*)(ws + 25 * MB);             // 1 MB

    k_prep0<<<NB * NN, 256, 0, stream>>>(x, adj, disH, adjH, deg, adjdisjk);
    k_prep1<<<(NB * NN) / 256, 256, 0, stream>>>(x, W1, b1, W2, b2, W3,
                                                 xWx, xWyb1, xWab2, xW3, xWT);
    // GEMM1: adj_dis_ik[i,j] = dot(dis_row_i, adj_row_j)   M=N=K=512
    {
        int waves = NB * 32 * 32;
        k_gemm_nt<<<waves / 4, 128, 0, stream>>>(disH, adjH, adjdisik,
                                                 32, 32, NN, NN,
                                                 (long)NN * NN, (long)NN * NN, (long)NN * NN);
    }
    // GEMM2: adjW[j,h] = dot(adj_row_j, xWT_row_h)   M=512, Nc=128, K=512
    {
        int waves = NB * 32 * 8;
        k_gemm_nt<<<waves / 4, 128, 0, stream>>>(adjH, xWT, adjW,
                                                 32, 8, NN, 128,
                                                 (long)NN * NN, (long)128 * NN, (long)NN * 128);
    }
    k_vj<<<(NB * NN * 64) / 256, 256, 0, stream>>>(xWyb1, xWab2, adjW, deg, adjdisjk,
                                                   W1, W2, vj, base2);
    k_m3<<<NB * NN, 64, 0, stream>>>(adjH, disH, adjdisik, deg, xWx, vj, W1, macc);
    k_m2m1<<<NB * NN, 64, 0, stream>>>(base2, macc, xW3, W2, W3, b3, out);
}
